// SoREX_60017872994472
// MI455X (gfx1250) — compile-verified, matrix operand order checked by reading, not run on hardware
//
#include <hip/hip_runtime.h>
#include <hip/hip_bf16.h>

#define NT       150000
#define NUSER    50000
#define NITEM    100000
#define DIM      128
#define NEG      10
#define ESOC     1600000
#define ECO      3200000
#define BATCH    512
#define MAX_UI   64
#define MAX_IU   50
#define NWALK    96          // W*HOP = 32*3
#define RMIX     0.7f

typedef __attribute__((ext_vector_type(16))) _Float16 v16h;
typedef __attribute__((ext_vector_type(8)))  _Float16 v8h;
typedef __attribute__((ext_vector_type(4)))  _Float16 h4v;
typedef __attribute__((ext_vector_type(8)))  float    v8f;
typedef int v4i __attribute__((vector_size(16)));   // matches builtin's V4i

#if __has_builtin(__builtin_amdgcn_global_load_async_to_lds_b128) && \
    __has_builtin(__builtin_amdgcn_s_wait_asynccnt)
#define USE_ASYNC_LDS 1
#endif

#ifdef USE_ASYNC_LDS
// 16B global -> LDS direct copy, tracked by ASYNCcnt (no VGPR round-trip).
// Prototype (from hipcc diagnostic): (v4i AS1* src, v4i AS3* dst, Imm off, Imm cpol)
__device__ __forceinline__ void async_copy16(const _Float16* g, _Float16* l) {
  __builtin_amdgcn_global_load_async_to_lds_b128(
      (__attribute__((address_space(1))) v4i*)(_Float16*)g,
      (__attribute__((address_space(3))) v4i*)l,
      /*imm offset=*/0, /*cpol=*/0);
}
#endif

// ---------------------------------------------------------------- utilities
__global__ void k_fill(float* __restrict__ p, long n, float v) {
  long i  = blockIdx.x * (long)blockDim.x + threadIdx.x;
  long st = (long)gridDim.x * blockDim.x;
  for (; i < n; i += st) p[i] = v;
}

__device__ __forceinline__ float wave_sum32(float v) {
  v += __shfl_xor(v, 1, 32);
  v += __shfl_xor(v, 2, 32);
  v += __shfl_xor(v, 4, 32);
  v += __shfl_xor(v, 8, 32);
  v += __shfl_xor(v, 16, 32);
  return v;
}

// ------------------------------------------------- soc edge softmax (by dst)
__global__ void k_edge_max(const int* __restrict__ dst, const float* __restrict__ w,
                           int* __restrict__ m, int E) {
  int e = blockIdx.x * blockDim.x + threadIdx.x;
  if (e >= E) return;
  atomicMax(&m[dst[e]], __float_as_int(w[e]));   // w in [0,1): int-ordered
}
__global__ void k_edge_sum(const int* __restrict__ dst, const float* __restrict__ w,
                           const int* __restrict__ m, float* __restrict__ s, int E) {
  int e = blockIdx.x * blockDim.x + threadIdx.x;
  if (e >= E) return;
  int d = dst[e];
  atomicAdd(&s[d], __expf(w[e] - __int_as_float(m[d])));
}
__global__ void k_edge_alpha(const int* __restrict__ dst, const float* __restrict__ w,
                             const int* __restrict__ m, const float* __restrict__ s,
                             float* __restrict__ alpha, int E) {
  int e = blockIdx.x * blockDim.x + threadIdx.x;
  if (e >= E) return;
  int d = dst[e];
  alpha[e] = __expf(w[e] - __int_as_float(m[d])) / (s[d] + 1e-16f);
}

// ---------------------------------------------------------------- SpMM (soc)
__global__ void k_spmm_edge(const int* __restrict__ src, const int* __restrict__ dst,
                            const float* __restrict__ alpha,
                            const float* __restrict__ xin, float* __restrict__ xout, int E) {
  long total  = (long)E * 32;
  long stride = (long)gridDim.x * blockDim.x;
  for (long t = blockIdx.x * (long)blockDim.x + threadIdx.x; t < total; t += stride) {
    long tn = t + stride;
    if (tn < total)
      __builtin_prefetch(xin + (long)src[tn >> 5] * DIM + (tn & 31) * 4, 0, 0);
    int e = (int)(t >> 5), q = (int)(t & 31);
    int sN = src[e], dN = dst[e];
    float a = alpha[e];
    float4 v = ((const float4*)xin)[(long)sN * 32 + q];
    float* o = xout + (long)dN * DIM + q * 4;
    atomicAdd(o + 0, a * v.x); atomicAdd(o + 1, a * v.y);
    atomicAdd(o + 2, a * v.z); atomicAdd(o + 3, a * v.w);
  }
}

// ------------------------------------------------------------- SpMM (co/sym)
__global__ void k_deg(const int* __restrict__ dst, float* __restrict__ deg, int E) {
  int e = blockIdx.x * blockDim.x + threadIdx.x;
  if (e >= E) return;
  atomicAdd(&deg[dst[e]], 1.0f);
}
// deg -> dinv (in place); deg >= 1 always (self loop)
__global__ void k_dinv(float* __restrict__ deg) {
  int n = blockIdx.x * blockDim.x + threadIdx.x;
  if (n >= NT) return;
  deg[n] = rsqrtf(deg[n]);
}
// self-loop term initializes the output: y[n] = dinv[n]^2 * x[n]
__global__ void k_co_self(const float* __restrict__ dinv, const float* __restrict__ xin,
                          float* __restrict__ xout) {
  long t = blockIdx.x * (long)blockDim.x + threadIdx.x;
  if (t >= (long)NT * 32) return;
  int n = (int)(t >> 5), q = (int)(t & 31);
  float dv = dinv[n];
  float w  = dv * dv;
  float4 v = ((const float4*)xin)[(long)n * 32 + q];
  v.x *= w; v.y *= w; v.z *= w; v.w *= w;
  ((float4*)xout)[(long)n * 32 + q] = v;
}
__global__ void k_spmm_sym(const int* __restrict__ src, const int* __restrict__ dst,
                           const float* __restrict__ dinv,
                           const float* __restrict__ xin, float* __restrict__ xout, int E) {
  long total  = (long)E * 32;
  long stride = (long)gridDim.x * blockDim.x;
  for (long t = blockIdx.x * (long)blockDim.x + threadIdx.x; t < total; t += stride) {
    long tn = t + stride;
    if (tn < total)
      __builtin_prefetch(xin + (long)src[tn >> 5] * DIM + (tn & 31) * 4, 0, 0);
    int e = (int)(t >> 5), q = (int)(t & 31);
    int sN = src[e], dN = dst[e];
    float a = dinv[sN] * dinv[dN];
    float4 v = ((const float4*)xin)[(long)sN * 32 + q];
    float* o = xout + (long)dN * DIM + q * 4;
    atomicAdd(o + 0, a * v.x); atomicAdd(o + 1, a * v.y);
    atomicAdd(o + 2, a * v.z); atomicAdd(o + 3, a * v.w);
  }
}

// ------------------------------------------- L2-normalize -> f16 (wave/row)
__global__ void k_norm(const float* __restrict__ x, _Float16* __restrict__ o, int rows) {
  int wid  = (blockIdx.x * blockDim.x + threadIdx.x) >> 5;
  int lane = threadIdx.x & 31;
  if (wid >= rows) return;
  float4 v = ((const float4*)x)[(long)wid * 32 + lane];
  float ss = wave_sum32(v.x * v.x + v.y * v.y + v.z * v.z + v.w * v.w);
  float rn = 1.0f / fmaxf(sqrtf(ss), 1e-12f);
  h4v h = { (_Float16)(v.x * rn), (_Float16)(v.y * rn),
            (_Float16)(v.z * rn), (_Float16)(v.w * rn) };
  ((h4v*)(o + (long)wid * DIM))[lane] = h;
}

// ------------------------- anchor: mean of soc_pad over item_users, L2, f16
__global__ void k_anchor(const float* __restrict__ soc_pad, const int* __restrict__ item_users,
                         _Float16* __restrict__ son) {
  int it   = (blockIdx.x * blockDim.x + threadIdx.x) >> 5;
  int lane = threadIdx.x & 31;
  if (it >= NITEM) return;
  float4 acc = {0.f, 0.f, 0.f, 0.f};
  int cnt = 0;
  for (int j = 0; j < MAX_IU; ++j) {
    int u = item_users[(long)it * MAX_IU + j];
    if (u >= 0) {
      ++cnt;
      float4 v = ((const float4*)soc_pad)[(long)u * 32 + lane];
      acc.x += v.x; acc.y += v.y; acc.z += v.z; acc.w += v.w;
    }
  }
  float inv = 1.0f / ((float)cnt + 1e-7f);
  acc.x *= inv; acc.y *= inv; acc.z *= inv; acc.w *= inv;
  float ss = wave_sum32(acc.x * acc.x + acc.y * acc.y + acc.z * acc.z + acc.w * acc.w);
  float rn = 1.0f / fmaxf(sqrtf(ss), 1e-12f);
  h4v h = { (_Float16)(acc.x * rn), (_Float16)(acc.y * rn),
            (_Float16)(acc.z * rn), (_Float16)(acc.w * rn) };
  ((h4v*)(son + ((long)(NUSER + it)) * DIM))[lane] = h;
}

// ----------------------------- negative sampling: threefry2x32 (key (0,42))
__device__ __forceinline__ unsigned rotl32(unsigned x, unsigned r) {
  return (x << r) | (x >> (32u - r));
}
__device__ float jax_uniform(long flat) {
  const unsigned k0 = 0u, k1 = 42u;
  const unsigned k2 = 0x1BD11BDAu ^ k0 ^ k1;
  const long H = (long)BATCH * NITEM / 2;
  unsigned x0, x1; bool hi;
  if (flat < H) { x0 = (unsigned)flat;       x1 = (unsigned)(flat + H); hi = false; }
  else          { x0 = (unsigned)(flat - H); x1 = (unsigned)flat;       hi = true;  }
  x0 += k0; x1 += k1;
  const unsigned rA[4] = {13u, 15u, 26u, 6u}, rB[4] = {17u, 29u, 16u, 24u};
#pragma unroll
  for (int i = 0; i < 4; ++i) { x0 += x1; x1 = rotl32(x1, rA[i]); x1 ^= x0; }
  x0 += k1; x1 += k2 + 1u;
#pragma unroll
  for (int i = 0; i < 4; ++i) { x0 += x1; x1 = rotl32(x1, rB[i]); x1 ^= x0; }
  x0 += k2; x1 += k0 + 2u;
#pragma unroll
  for (int i = 0; i < 4; ++i) { x0 += x1; x1 = rotl32(x1, rA[i]); x1 ^= x0; }
  x0 += k0; x1 += k1 + 3u;
#pragma unroll
  for (int i = 0; i < 4; ++i) { x0 += x1; x1 = rotl32(x1, rB[i]); x1 ^= x0; }
  x0 += k1; x1 += k2 + 4u;
#pragma unroll
  for (int i = 0; i < 4; ++i) { x0 += x1; x1 = rotl32(x1, rA[i]); x1 ^= x0; }
  x0 += k2; x1 += k0 + 5u;
  unsigned bits = hi ? x1 : x0;
  return __uint_as_float((bits >> 9) | 0x3f800000u) - 1.0f;
}

__global__ void k_mark(const int* __restrict__ user_items, unsigned* __restrict__ bitmap) {
  int t = blockIdx.x * blockDim.x + threadIdx.x;
  if (t >= BATCH * MAX_UI) return;
  int b = t / MAX_UI;
  int it = user_items[t];
  if (it >= 0 && it < NITEM)
    atomicOr(&bitmap[(long)b * (NITEM / 32) + (it >> 5)], 1u << (it & 31));
}

__global__ __launch_bounds__(256) void k_topk(const unsigned* __restrict__ bitmap,
                                              const int* __restrict__ n_id,
                                              int* __restrict__ cand) {
  __shared__ float lv[256 * NEG];
  __shared__ int   li[256 * NEG];
  int b = blockIdx.x, tid = threadIdx.x;
  float bv[NEG]; int bi[NEG];
#pragma unroll
  for (int j = 0; j < NEG; ++j) { bv[j] = -2.0f; bi[j] = 0x7fffffff; }
  const unsigned* bm = bitmap + (long)b * (NITEM / 32);
  for (int i = tid; i < NITEM; i += 256) {
    if (bm[i >> 5] & (1u << (i & 31))) continue;     // item already interacted
    float u = jax_uniform((long)b * NITEM + i);
    if (u > bv[NEG - 1] || (u == bv[NEG - 1] && i < bi[NEG - 1])) {
      bv[NEG - 1] = u; bi[NEG - 1] = i;
#pragma unroll
      for (int j = NEG - 1; j > 0; --j) {
        bool sw = (bv[j] > bv[j - 1]) || (bv[j] == bv[j - 1] && bi[j] < bi[j - 1]);
        if (sw) {
          float tv = bv[j]; bv[j] = bv[j - 1]; bv[j - 1] = tv;
          int   ti = bi[j]; bi[j] = bi[j - 1]; bi[j - 1] = ti;
        }
      }
    }
  }
#pragma unroll
  for (int j = 0; j < NEG; ++j) { lv[tid * NEG + j] = bv[j]; li[tid * NEG + j] = bi[j]; }
  __syncthreads();
  // pairwise tree merge of sorted-descending 10-lists: 8 steps instead of a
  // 25K-iteration serial scan.  Merging thread owns slots {tid, tid+stride}.
  for (int stride = 1; stride < 256; stride <<= 1) {
    if ((tid & (2 * stride - 1)) == 0) {
      int ba = tid * NEG, bb = (tid + stride) * NEG;
      float ov[NEG]; int oi[NEG];
      int pa = 0, pb = 0;
#pragma unroll
      for (int j = 0; j < NEG; ++j) {
        float va = (pa < NEG) ? lv[ba + pa] : -4.0f;
        int   ia = (pa < NEG) ? li[ba + pa] : 0x7fffffff;
        float vb = (pb < NEG) ? lv[bb + pb] : -4.0f;
        int   ib = (pb < NEG) ? li[bb + pb] : 0x7fffffff;
        bool ta = (va > vb) || (va == vb && ia < ib);
        ov[j] = ta ? va : vb;
        oi[j] = ta ? ia : ib;
        if (ta) ++pa; else ++pb;
      }
#pragma unroll
      for (int j = 0; j < NEG; ++j) { lv[ba + j] = ov[j]; li[ba + j] = oi[j]; }
    }
    __syncthreads();
  }
  if (tid == 0) {
    cand[b * (NEG + 1)] = n_id[b * 2 + 1];           // positive item (global id)
    for (int j = 0; j < NEG; ++j)
      cand[b * (NEG + 1) + 1 + j] = NUSER + li[j];
  }
}

// ------------------------------- final similarity: WMMA f16 -> f32 per batch
// block = 64 threads (2 waves). wave0: co-graph sims, wave1: soc sims.
// Each wave accumulates ONE 16x16 f32 tile over 6 walk-tiles x 4 K-steps,
// since mean over 96 walk columns commutes with the accumulation.
__global__ __launch_bounds__(64) void k_sim(const int* __restrict__ cand,
                                            const int* __restrict__ walks,
                                            const _Float16* __restrict__ con,
                                            const _Float16* __restrict__ son,
                                            float* __restrict__ out) {
  __shared__ _Float16 sA[2][16][DIM];       // cand rows (padded to 16), co/soc
  __shared__ _Float16 sB[2][NWALK][DIM];    // walk rows, co/soc
  __shared__ float ssum[2][16];
  int b = blockIdx.x, tid = threadIdx.x;
  const _Float16* srcp[2] = { con, son };

#ifdef USE_ASYNC_LDS
  // zero the pad rows (11..15) with plain LDS stores
  for (int w = tid; w < 2 * (16 - (NEG + 1)) * (DIM / 2); w += 64) {
    int t = w / ((16 - (NEG + 1)) * (DIM / 2));
    int r = (NEG + 1) + (w / (DIM / 2)) % (16 - (NEG + 1));
    int c = w % (DIM / 2);
    ((unsigned*)&sA[t][r][0])[c] = 0u;
  }
  // candidate rows: 16B async global->LDS copies (ASYNCcnt)
  for (int w = tid; w < 2 * 16 * (DIM / 8); w += 64) {
    int t = w / (16 * (DIM / 8));
    int r = (w / (DIM / 8)) % 16;
    int c = w % (DIM / 8);
    if (r < NEG + 1) {
      long node = cand[b * (NEG + 1) + r];
      async_copy16(srcp[t] + node * DIM + c * 8, &sA[t][r][c * 8]);
    }
  }
  // walk rows
  for (int w = tid; w < 2 * NWALK * (DIM / 8); w += 64) {
    int t = w / (NWALK * (DIM / 8));
    int r = (w / (DIM / 8)) % NWALK;
    int c = w % (DIM / 8);
    long node = walks[b * NWALK + r];
    async_copy16(srcp[t] + node * DIM + c * 8, &sB[t][r][c * 8]);
  }
  __builtin_amdgcn_s_wait_asynccnt(0);
#else
  for (int w = tid; w < 2 * 16 * (DIM / 2); w += 64) {
    int t = w / (16 * (DIM / 2));
    int r = (w / (DIM / 2)) % 16;
    int c = w % (DIM / 2);
    unsigned val = 0u;
    if (r < NEG + 1) {
      long node = cand[b * (NEG + 1) + r];
      val = ((const unsigned*)srcp[t])[node * (DIM / 2) + c];
    }
    ((unsigned*)&sA[t][r][0])[c] = val;
  }
  for (int w = tid; w < 2 * NWALK * (DIM / 2); w += 64) {
    int t = w / (NWALK * (DIM / 2));
    int r = (w / (DIM / 2)) % NWALK;
    int c = w % (DIM / 2);
    long node = walks[b * NWALK + r];
    ((unsigned*)&sB[t][r][0])[c] = ((const unsigned*)srcp[t])[node * (DIM / 2) + c];
  }
#endif
  __syncthreads();

  int wave = tid >> 5;          // 0 = co, 1 = soc
  int lane = tid & 31;
  int g    = lane >> 4;         // half-wave group
  int l16  = lane & 15;

  v8f acc = {};                 // 16x16 f32 accumulator tile
  for (int tile = 0; tile < NWALK / 16; ++tile) {
#pragma unroll
    for (int k = 0; k < DIM / 32; ++k) {
      // 16-bit A/B VGPR layout: lane holds K = kb..kb+7 and kb+16..kb+23,
      // where kb = 32*k + 8*g  (half-wave K split).
      int kb = k * 32 + g * 8;
      v8h alo = *(const v8h*)&sA[wave][l16][kb];
      v8h ahi = *(const v8h*)&sA[wave][l16][kb + 16];
      v8h blo = *(const v8h*)&sB[wave][tile * 16 + l16][kb];
      v8h bhi = *(const v8h*)&sB[wave][tile * 16 + l16][kb + 16];
      v16h a, bm;
#pragma unroll
      for (int i = 0; i < 8; ++i) {
        a[i] = alo[i];  a[i + 8] = ahi[i];
        bm[i] = blo[i]; bm[i + 8] = bhi[i];
      }
      acc = __builtin_amdgcn_wmma_f32_16x16x32_f16(
          /*neg_a=*/false, a, /*neg_b=*/false, bm,
          /*c_mod=*/(short)0, acc, /*reuse_a=*/false, /*reuse_b=*/false);
    }
  }

  // D layout: lane -> N=l16, VGPR r -> M = r + 8*g. Sum over N (16 lanes).
#pragma unroll
  for (int r = 0; r < 8; ++r) {
    float v = acc[r];
    v += __shfl_xor(v, 1, 32);
    v += __shfl_xor(v, 2, 32);
    v += __shfl_xor(v, 4, 32);
    v += __shfl_xor(v, 8, 32);
    if (l16 == 0) ssum[wave][g * 8 + r] = v;
  }
  __syncthreads();
  if (tid < NEG + 1) {
    float mc = ssum[0][tid] * (1.0f / NWALK);   // mean dot over 96 walks
    float ms = ssum[1][tid] * (1.0f / NWALK);
    out[b * (NEG + 1) + tid] =
        RMIX * (mc + 1.0f) * 0.5f + (1.0f - RMIX) * (ms + 1.0f) * 0.5f;
  }
}

// ---------------------------------------------------------------- dispatcher
extern "C" void kernel_launch(void* const* d_in, const int* in_sizes, int n_in,
                              void* d_out, int out_size, void* d_ws, size_t ws_size,
                              hipStream_t stream) {
  (void)in_sizes; (void)n_in; (void)out_size; (void)ws_size;
  const float* id_emb     = (const float*)d_in[0];
  const float* soc_id_emb = (const float*)d_in[1];
  const float* soc_w      = (const float*)d_in[2];
  const int*   soc_ei     = (const int*)d_in[3];   // (2, ESOC): src row, dst row
  const int*   co_ei      = (const int*)d_in[4];   // (2, ECO)
  const int*   n_id       = (const int*)d_in[5];
  const int*   user_items = (const int*)d_in[6];
  const int*   item_users = (const int*)d_in[7];
  const int*   walks      = (const int*)d_in[8];
  float* out = (float*)d_out;

  const int* soc_src = soc_ei;           const int* soc_dst = soc_ei + ESOC;
  const int* co_src  = co_ei;            const int* co_dst  = co_ei + ECO;

  char* ws = (char*)d_ws; size_t off = 0;
  auto carve = [&](size_t bytes) -> void* {
    void* p = ws + off; off = (off + bytes + 255) & ~(size_t)255; return p;
  };
  float*    alpha = (float*)carve(sizeof(float) * ESOC);
  int*      mbuf  = (int*)  carve(sizeof(int)   * NT);
  float*    sbuf  = (float*)carve(sizeof(float) * NT);
  float*    xA    = (float*)carve(sizeof(float) * (long)(NT + 1) * DIM);  // soc L1
  float*    xB    = (float*)carve(sizeof(float) * (long)(NT + 1) * DIM);  // soc L2 (+pad)
  float*    deg   = (float*)carve(sizeof(float) * NT);                    // -> dinv
  float*    yA    = (float*)carve(sizeof(float) * (long)NT * DIM);        // co L1
  float*    yB    = (float*)carve(sizeof(float) * (long)NT * DIM);        // co L2
  _Float16* con   = (_Float16*)carve(sizeof(_Float16) * (long)(NT + 1) * DIM);
  _Float16* son   = (_Float16*)carve(sizeof(_Float16) * (long)(NT + 1) * DIM);
  unsigned* bitmap= (unsigned*)carve(sizeof(unsigned) * (long)BATCH * (NITEM / 32));
  int*      cand  = (int*)carve(sizeof(int) * BATCH * (NEG + 1));

  auto fill = [&](float* p, long n, float v) {
    long blk = (n + 255) / 256; if (blk > 32768) blk = 32768;
    k_fill<<<(int)blk, 256, 0, stream>>>(p, n, v);
  };

  // --- init scratch
  fill(xA, (long)(NT + 1) * DIM, 0.0f);
  fill(xB, (long)(NT + 1) * DIM, 0.0f);
  fill((float*)mbuf, NT, 0.0f);                 // bits(0.0f) == 0
  fill(sbuf, NT, 0.0f);
  fill(deg, NT, 1.0f);                          // self-loop contributes 1
  fill((float*)bitmap, (long)BATCH * (NITEM / 32), 0.0f);
  fill((float*)(con + (long)NT * DIM), DIM / 2, 0.0f);   // pad rows = 0
  fill((float*)(son + (long)NT * DIM), DIM / 2, 0.0f);

  // --- soc: edge softmax + 2-layer propagation
  int gE = (ESOC + 255) / 256;
  k_edge_max  <<<gE, 256, 0, stream>>>(soc_dst, soc_w, mbuf, ESOC);
  k_edge_sum  <<<gE, 256, 0, stream>>>(soc_dst, soc_w, mbuf, sbuf, ESOC);
  k_edge_alpha<<<gE, 256, 0, stream>>>(soc_dst, soc_w, mbuf, sbuf, alpha, ESOC);
  k_spmm_edge<<<32768, 256, 0, stream>>>(soc_src, soc_dst, alpha, soc_id_emb, xA, ESOC);
  k_spmm_edge<<<32768, 256, 0, stream>>>(soc_src, soc_dst, alpha, xA, xB, ESOC);

  // --- co: degree + 2-layer symmetric propagation (self-loop via init)
  int gC = (ECO + 255) / 256;
  k_deg <<<gC, 256, 0, stream>>>(co_dst, deg, ECO);
  k_dinv<<<(NT + 255) / 256, 256, 0, stream>>>(deg);
  int gSelf = (int)(((long)NT * 32 + 255) / 256);
  k_co_self <<<gSelf, 256, 0, stream>>>(deg, id_emb, yA);
  k_spmm_sym<<<32768, 256, 0, stream>>>(co_src, co_dst, deg, id_emb, yA, ECO);
  k_co_self <<<gSelf, 256, 0, stream>>>(deg, yA, yB);
  k_spmm_sym<<<32768, 256, 0, stream>>>(co_src, co_dst, deg, yA, yB, ECO);

  // --- normalize to f16 (co: all nodes; soc users; soc items via anchor)
  k_norm<<<(int)(((long)NT * 32 + 255) / 256), 256, 0, stream>>>(yB, con, NT);
  k_norm<<<(int)(((long)NUSER * 32 + 255) / 256), 256, 0, stream>>>(xB, son, NUSER);
  k_anchor<<<(int)(((long)NITEM * 32 + 255) / 256), 256, 0, stream>>>(xB, item_users, son);

  // --- negative sampling
  k_mark<<<(BATCH * MAX_UI + 255) / 256, 256, 0, stream>>>(user_items, bitmap);
  k_topk<<<BATCH, 256, 0, stream>>>(bitmap, n_id, cand);

  // --- WMMA similarity + mix
  k_sim<<<BATCH, 64, 0, stream>>>(cand, walks, con, son, out);
}
